// OptimizedMultiHeadAttention_90598040142355
// MI455X (gfx1250) — matrix operand order repaired and checked by reading, not compiled, not run on hardware
//
#include <hip/hip_runtime.h>

#define DMODEL 512
#define NHEADS 8
#define DK 64
#define SEQ 4096
#define BATCH 2
#define MTOK (BATCH * SEQ) // 8192

typedef __attribute__((ext_vector_type(16))) __bf16 v16bf;
typedef __attribute__((ext_vector_type(8)))  __bf16 v8bf;
typedef __attribute__((ext_vector_type(8)))  float  v8f;
typedef __attribute__((ext_vector_type(4)))  float  v4f;

static __device__ inline v8f wmma_bf16(v16bf a, v16bf b, v8f c) {
  // (neg_a, A, neg_b, B, c_mod, C, reuse_a, reuse_b)
  return __builtin_amdgcn_wmma_f32_16x16x32_bf16(false, a, false, b, (short)0, c, false, false);
}

// A-fragment (16x32 bf16, M x K) from row-major [rows, ld] starting at (row0,k0).
// Layout: lane<16 -> row=lane, K0..15; lane>=16 -> row=lane-16, K16..31 (contiguous, matches B)
static __device__ inline v16bf load_afrag_g(const __bf16* __restrict__ base, int ld, int lane) {
  int row  = lane & 15;
  int koff = (lane & 16) ? 16 : 0;
  const __bf16* p = base + (size_t)row * ld + koff;
  union { v16bf v; v8bf h[2]; } u;
  u.h[0] = *reinterpret_cast<const v8bf*>(p);
  u.h[1] = *reinterpret_cast<const v8bf*>(p + 8);
  return u.v;
}

// B-fragment (32x16 bf16, K x N) for Y = X * W^T:  B[k][n] = W[n][k], W row-major [N, ld].
// Layout: lane<16 -> N=lane, K0..15 ; lane>=16 -> N=lane-16, K16..31 (all contiguous per lane)
static __device__ inline v16bf load_bfrag_g(const __bf16* __restrict__ base, int ld, int lane) {
  int col  = lane & 15;
  int koff = (lane & 16) ? 16 : 0;
  return *reinterpret_cast<const v16bf*>(base + (size_t)col * ld + koff);
}

// ---------------------------------------------------------------- converts
__global__ void f32_to_bf16_kernel(const float* __restrict__ src, __bf16* __restrict__ dst, int n) {
  int i = blockIdx.x * blockDim.x + threadIdx.x;
  if (i < n) dst[i] = (__bf16)src[i];
}

// ---------------------------------------------------------------- LayerNorm (one wave per token)
__global__ __launch_bounds__(128) void ln_bf16_kernel(
    const float* __restrict__ x, const float* __restrict__ gamma,
    const float* __restrict__ beta, __bf16* __restrict__ h) {
  int lane = threadIdx.x & 31;
  int wid  = threadIdx.x >> 5;
  int token = blockIdx.x * 4 + wid;
  const float* xr = x + (size_t)token * DMODEL + lane * 16;
  float v[16];
#pragma unroll
  for (int i = 0; i < 4; ++i) {
    v4f t = *reinterpret_cast<const v4f*>(xr + 4 * i);
#pragma unroll
    for (int j = 0; j < 4; ++j) v[4 * i + j] = t[j];
  }
  float s = 0.f, ss = 0.f;
#pragma unroll
  for (int i = 0; i < 16; ++i) { s += v[i]; ss += v[i] * v[i]; }
  for (int m = 1; m <= 16; m <<= 1) { s += __shfl_xor(s, m, 32); ss += __shfl_xor(ss, m, 32); }
  float mu   = s * (1.0f / DMODEL);
  float var  = ss * (1.0f / DMODEL) - mu * mu;
  float rstd = rsqrtf(var + 1e-5f);
  union { __bf16 o[16]; v8bf v8[2]; } ou;
#pragma unroll
  for (int i = 0; i < 16; ++i) {
    int c = lane * 16 + i;
    ou.o[i] = (__bf16)((v[i] - mu) * rstd * gamma[c] + beta[c]);
  }
  __bf16* hr = h + (size_t)token * DMODEL + lane * 16;
  *reinterpret_cast<v8bf*>(hr)     = ou.v8[0];
  *reinterpret_cast<v8bf*>(hr + 8) = ou.v8[1];
}

// ---------------------------------------------------------------- QKV projections
// mode 0 -> Q [b][h][s][d], mode 1 -> K [b][h][s][d], mode 2 -> V transposed [b][h][d][s]
__global__ __launch_bounds__(128) void qkv_gemm_kernel(
    const __bf16* __restrict__ h,
    const __bf16* __restrict__ wq, const __bf16* __restrict__ wk, const __bf16* __restrict__ wvp,
    __bf16* __restrict__ Qb, __bf16* __restrict__ Kb, __bf16* __restrict__ Vt) {
  int mode = blockIdx.z;
  const __bf16* w = (mode == 0) ? wq : (mode == 1 ? wk : wvp);
  int lane = threadIdx.x & 31;
  int wid  = threadIdx.x >> 5;
  int m0 = blockIdx.x * 128 + (wid >> 1) * 64;
  int n0 = blockIdx.y * 128 + (wid & 1) * 64;
  v8f acc[4][4] = {};
  for (int k0 = 0; k0 < DMODEL; k0 += 32) {
    v16bf a[4], bf[4];
#pragma unroll
    for (int i = 0; i < 4; ++i)
      a[i] = load_afrag_g(h + (size_t)(m0 + 16 * i) * DMODEL + k0, DMODEL, lane);
#pragma unroll
    for (int j = 0; j < 4; ++j)
      bf[j] = load_bfrag_g(w + (size_t)(n0 + 16 * j) * DMODEL + k0, DMODEL, lane);
#pragma unroll
    for (int i = 0; i < 4; ++i)
#pragma unroll
      for (int j = 0; j < 4; ++j)
        acc[i][j] = wmma_bf16(a[i], bf[j], acc[i][j]);
  }
  int c  = lane & 15;
  int r0 = (lane & 16) ? 8 : 0;
#pragma unroll
  for (int i = 0; i < 4; ++i)
#pragma unroll
    for (int j = 0; j < 4; ++j) {
      int n = n0 + 16 * j + c;
      int head = n >> 6, d = n & 63;
#pragma unroll
      for (int e = 0; e < 8; ++e) {
        int m = m0 + 16 * i + r0 + e;
        int bb = m >> 12, s = m & (SEQ - 1);
        __bf16 val = (__bf16)acc[i][j][e];
        if (mode == 2)
          Vt[(size_t)((bb * NHEADS + head) * DK + d) * SEQ + s] = val;
        else if (mode == 0)
          Qb[(size_t)((bb * NHEADS + head) * SEQ + s) * DK + d] = val;
        else
          Kb[(size_t)((bb * NHEADS + head) * SEQ + s) * DK + d] = val;
      }
    }
}

// ---------------------------------------------------------------- flash attention
// one wave per 16-query tile per (b,h); online softmax; P converted C-layout -> A-layout via LDS
__global__ __launch_bounds__(128) void flash_attn_kernel(
    const __bf16* __restrict__ Qb, const __bf16* __restrict__ Kb,
    const __bf16* __restrict__ Vt, __bf16* __restrict__ att) {
  __shared__ __align__(32) __bf16 ldsP[4][16 * 40]; // 40-el row stride -> 16B-aligned chunks
  int lane = threadIdx.x & 31;
  int wid  = threadIdx.x >> 5;
  int b  = blockIdx.z;
  int hh = blockIdx.y;
  int q0 = (blockIdx.x * 4 + wid) * 16;
  const __bf16* Qbase = Qb + (size_t)(b * NHEADS + hh) * SEQ * DK;
  const __bf16* Kbase = Kb + (size_t)(b * NHEADS + hh) * SEQ * DK;
  const __bf16* Vbase = Vt + (size_t)(b * NHEADS + hh) * DK * SEQ;
  v16bf qa0 = load_afrag_g(Qbase + (size_t)q0 * DK + 0,  DK, lane);
  v16bf qa1 = load_afrag_g(Qbase + (size_t)q0 * DK + 32, DK, lane);
  float mrow[8], lrow[8];
  v8f O[4] = {};
#pragma unroll
  for (int e = 0; e < 8; ++e) { mrow[e] = -1e30f; lrow[e] = 0.f; }
  __bf16* P = ldsP[wid];
  int c     = lane & 15;
  int r0    = (lane & 16) ? 8 : 0;
  int koffA = (lane & 16) ? 16 : 0;
  int koffB = (lane & 16) ? 16 : 0;
  for (int j0 = 0; j0 < SEQ; j0 += 32) {
    v8f s0 = {}, s1 = {};
    v16bf b00 = load_bfrag_g(Kbase + (size_t)j0 * DK,             DK, lane);
    v16bf b01 = load_bfrag_g(Kbase + (size_t)j0 * DK + 32,       DK, lane);
    v16bf b10 = load_bfrag_g(Kbase + (size_t)(j0 + 16) * DK,      DK, lane);
    v16bf b11 = load_bfrag_g(Kbase + (size_t)(j0 + 16) * DK + 32, DK, lane);
    s0 = wmma_bf16(qa0, b00, s0);
    s0 = wmma_bf16(qa1, b01, s0);
    s1 = wmma_bf16(qa0, b10, s1);
    s1 = wmma_bf16(qa1, b11, s1);
    if (j0 + 32 < SEQ)
      __builtin_prefetch(Kbase + (size_t)(j0 + 32) * DK + lane * 8, 0, 0);
    float corr[8], p0[8], p1[8];
#pragma unroll
    for (int e = 0; e < 8; ++e) {
      float a  = s0[e] * 0.125f;             // 1/sqrt(64)
      float cc = s1[e] * 0.125f;
      float mx = fmaxf(a, cc);               // row max over 16 lanes of the half-wave
      mx = fmaxf(mx, __shfl_xor(mx, 1, 32));
      mx = fmaxf(mx, __shfl_xor(mx, 2, 32));
      mx = fmaxf(mx, __shfl_xor(mx, 4, 32));
      mx = fmaxf(mx, __shfl_xor(mx, 8, 32));
      float nm = fmaxf(mrow[e], mx);
      corr[e] = __expf(mrow[e] - nm);
      p0[e]   = __expf(a - nm);
      p1[e]   = __expf(cc - nm);
      float sum = p0[e] + p1[e];
      sum += __shfl_xor(sum, 1, 32);
      sum += __shfl_xor(sum, 2, 32);
      sum += __shfl_xor(sum, 4, 32);
      sum += __shfl_xor(sum, 8, 32);
      lrow[e] = lrow[e] * corr[e] + sum;
      mrow[e] = nm;
    }
#pragma unroll
    for (int t = 0; t < 4; ++t)
#pragma unroll
      for (int e = 0; e < 8; ++e) O[t][e] *= corr[e];
    // C-layout -> LDS (row-major 16x32, bf16); per-wave private region, DS ops in-order
#pragma unroll
    for (int e = 0; e < 8; ++e) {
      P[(r0 + e) * 40 + c]      = (__bf16)p0[e];
      P[(r0 + e) * 40 + c + 16] = (__bf16)p1[e];
    }
    union { v16bf v; v8bf h2[2]; } pu;
    pu.h2[0] = *reinterpret_cast<const v8bf*>(P + (lane & 15) * 40 + koffA);
    pu.h2[1] = *reinterpret_cast<const v8bf*>(P + (lane & 15) * 40 + koffA + 8);
#pragma unroll
    for (int t = 0; t < 4; ++t) {
      v16bf vb = *reinterpret_cast<const v16bf*>(
          Vbase + (size_t)(16 * t + c) * SEQ + j0 + koffB);
      O[t] = wmma_bf16(pu.v, vb, O[t]);
    }
  }
  float inv[8];
#pragma unroll
  for (int e = 0; e < 8; ++e) inv[e] = 1.0f / lrow[e];
#pragma unroll
  for (int t = 0; t < 4; ++t)
#pragma unroll
    for (int e = 0; e < 8; ++e) {
      size_t row = (size_t)b * SEQ + q0 + r0 + e;
      att[row * DMODEL + hh * DK + 16 * t + c] = (__bf16)(O[t][e] * inv[e]);
    }
}

// ---------------------------------------------------------------- out = att @ Wo^T + b_o + x
__global__ __launch_bounds__(128) void out_proj_kernel(
    const __bf16* __restrict__ att, const __bf16* __restrict__ wo,
    const float* __restrict__ bo, const float* __restrict__ x,
    float* __restrict__ out) {
  int lane = threadIdx.x & 31;
  int wid  = threadIdx.x >> 5;
  int m0 = blockIdx.x * 128 + (wid >> 1) * 64;
  int n0 = blockIdx.y * 128 + (wid & 1) * 64;
  v8f acc[4][4] = {};
  for (int k0 = 0; k0 < DMODEL; k0 += 32) {
    v16bf a[4], bf[4];
#pragma unroll
    for (int i = 0; i < 4; ++i)
      a[i] = load_afrag_g(att + (size_t)(m0 + 16 * i) * DMODEL + k0, DMODEL, lane);
#pragma unroll
    for (int j = 0; j < 4; ++j)
      bf[j] = load_bfrag_g(wo + (size_t)(n0 + 16 * j) * DMODEL + k0, DMODEL, lane);
#pragma unroll
    for (int i = 0; i < 4; ++i)
#pragma unroll
      for (int j = 0; j < 4; ++j)
        acc[i][j] = wmma_bf16(a[i], bf[j], acc[i][j]);
  }
  int c  = lane & 15;
  int r0 = (lane & 16) ? 8 : 0;
#pragma unroll
  for (int j = 0; j < 4; ++j) {
    int n = n0 + 16 * j + c;
    float bb = bo[n];
#pragma unroll
    for (int i = 0; i < 4; ++i)
#pragma unroll
      for (int e = 0; e < 8; ++e) {
        size_t idx = (size_t)(m0 + 16 * i + r0 + e) * DMODEL + n;
        out[idx] = acc[i][j][e] + bb + x[idx];
      }
  }
}

// ---------------------------------------------------------------- launch
extern "C" void kernel_launch(void* const* d_in, const int* in_sizes, int n_in,
                              void* d_out, int out_size, void* d_ws, size_t ws_size,
                              hipStream_t stream) {
  (void)in_sizes; (void)n_in; (void)out_size; (void)ws_size;
  const float* x     = (const float*)d_in[0];
  const float* w_q   = (const float*)d_in[1];
  const float* w_k   = (const float*)d_in[2];
  const float* w_v   = (const float*)d_in[3];
  const float* w_o   = (const float*)d_in[4];
  const float* b_o   = (const float*)d_in[5];
  const float* gamma = (const float*)d_in[6];
  const float* beta  = (const float*)d_in[7];

  char* ws = (char*)d_ws;
  size_t off = 0;
  __bf16* hbuf = (__bf16*)(ws + off); off += (size_t)MTOK * DMODEL * 2;   // 8 MB (reused as 'attended')
  __bf16* wqb  = (__bf16*)(ws + off); off += (size_t)DMODEL * DMODEL * 2; // 512 KB each
  __bf16* wkb  = (__bf16*)(ws + off); off += (size_t)DMODEL * DMODEL * 2;
  __bf16* wvb  = (__bf16*)(ws + off); off += (size_t)DMODEL * DMODEL * 2;
  __bf16* wob  = (__bf16*)(ws + off); off += (size_t)DMODEL * DMODEL * 2;
  __bf16* Qb   = (__bf16*)(ws + off); off += (size_t)MTOK * DMODEL * 2;
  __bf16* Kb   = (__bf16*)(ws + off); off += (size_t)MTOK * DMODEL * 2;
  __bf16* Vt   = (__bf16*)(ws + off); off += (size_t)MTOK * DMODEL * 2;   // total ~35.7 MB

  const int WN = DMODEL * DMODEL; // 262144
  f32_to_bf16_kernel<<<(WN + 255) / 256, 256, 0, stream>>>(w_q, wqb, WN);
  f32_to_bf16_kernel<<<(WN + 255) / 256, 256, 0, stream>>>(w_k, wkb, WN);
  f32_to_bf16_kernel<<<(WN + 255) / 256, 256, 0, stream>>>(w_v, wvb, WN);
  f32_to_bf16_kernel<<<(WN + 255) / 256, 256, 0, stream>>>(w_o, wob, WN);

  ln_bf16_kernel<<<MTOK / 4, 128, 0, stream>>>(x, gamma, beta, hbuf);

  qkv_gemm_kernel<<<dim3(MTOK / 128, DMODEL / 128, 3), 128, 0, stream>>>(
      hbuf, wqb, wkb, wvb, Qb, Kb, Vt);

  // hbuf no longer needed -> reuse it as 'attended'
  flash_attn_kernel<<<dim3(SEQ / 64, NHEADS, BATCH), 128, 0, stream>>>(Qb, Kb, Vt, hbuf);

  out_proj_kernel<<<dim3(MTOK / 128, DMODEL / 128), 128, 0, stream>>>(
      hbuf, wob, b_o, x, (float*)d_out);
}
